// GraphNet_3521873183412
// MI455X (gfx1250) — matrix operand order repaired and checked by reading, not compiled, not run on hardware
//
#include <hip/hip_runtime.h>
#include <hip/hip_bf16.h>

typedef __attribute__((ext_vector_type(16))) _Float16 v16h;
typedef __attribute__((ext_vector_type(8)))  float    v8f;

namespace {
constexpr int BSZ = 4, NNODE = 256, DINC = 4, DM = 64, NHD = 4, NLAY = 2, DHD = 16;
constexpr int ROWS = BSZ * NNODE;           // 1024 tokens
constexpr float EPSV  = 1e-5f;
constexpr float SLOPE = 0.01f;

// ---------------- workspace byte offsets (all written before read) ----------------
constexpr size_t OFF_E     = 0;                       // 1,048,576 f32 edge exp (reused as attn scores)
constexpr size_t OFF_PROBS = OFF_E     + 4194304;     // 1,048,576 f16 attn probs
constexpr size_t OFF_DEN   = OFF_PROBS + 2097152;     // 4096  f32 per-(head,b,i) softmax denom
constexpr size_t OFF_G     = OFF_DEN   + 16384;       // 4096  f32 folded aggregate (B,N,DIN)
constexpr size_t OFF_S     = OFF_G     + 16384;       // 1024  f32 weight sums
constexpr size_t OFF_H     = OFF_S     + 4096;        // 65536 f32 pre-BN features
constexpr size_t OFF_X     = OFF_H     + 262144;      // 65536 f32 running activations
constexpr size_t OFF_XH    = OFF_X     + 262144;      // 65536 f16 activations (WMMA A operand)
constexpr size_t OFF_BN    = OFF_XH    + 131072;      // 64+64 f32 BN mean|var
constexpr size_t OFF_CW    = OFF_BN    + 512;         // 32 f32 folded score weights
constexpr size_t OFF_W16   = OFF_CW    + 128;         // 65536 f16 converted weights
constexpr size_t OFF_Q     = OFF_W16   + 131072;      // 65536 f16
constexpr size_t OFF_K     = OFF_Q     + 131072;      // 65536 f16
constexpr size_t OFF_V     = OFF_K     + 131072;      // 65536 f16
constexpr size_t OFF_O     = OFF_V     + 131072;      // 65536 f16
constexpr size_t OFF_PROJ  = OFF_O     + 131072;      // 65536 f32
constexpr size_t OFF_HH    = OFF_PROJ  + 262144;      // 131072 f16 FF hidden
// f16 weight sub-offsets (halves)
constexpr size_t WQ16 = 0, WK16 = 8192, WV16 = 16384, WO16 = 24576, WF1 = 32768, WF2 = 49152;
}

// ------------------------------ wave32 reductions ------------------------------
__device__ __forceinline__ float wave_sum(float v) {
#pragma unroll
  for (int o = 16; o > 0; o >>= 1) v += __shfl_xor(v, o, 32);
  return v;
}
__device__ __forceinline__ float wave_max(float v) {
#pragma unroll
  for (int o = 16; o > 0; o >>= 1) v = fmaxf(v, __shfl_xor(v, o, 32));
  return v;
}

// ---------------- WMMA fragment loaders (contiguous k = 16*(lane/16) + s, wave32) ----------------
__device__ __forceinline__ v16h frag_a(const _Float16* __restrict__ A, int lda,
                                       int m0, int k0, int lane) {
  const int row = m0 + (lane & 15);
  const int kb  = k0 + ((lane & 16) ? 16 : 0);
  const _Float16* p = A + (size_t)row * lda + kb;
  v16h f;
#pragma unroll
  for (int t = 0; t < 8; ++t) f[t] = p[t];          // K = kb .. kb+7
#pragma unroll
  for (int t = 0; t < 8; ++t) f[8 + t] = p[8 + t];  // K = kb+8 .. kb+15
  return f;
}
__device__ __forceinline__ v16h frag_b(const _Float16* __restrict__ Bm, int ldb,
                                       int k0, int n0, int lane) {
  const int col = n0 + (lane & 15);
  const int kb  = k0 + ((lane & 16) ? 16 : 0);
  const _Float16* p = Bm + (size_t)kb * ldb + col;
  v16h f;
#pragma unroll
  for (int t = 0; t < 16; ++t) f[t] = p[(size_t)t * ldb];
  return f;
}

// ------------------- generic 16x16-tile WMMA GEMM: Out = act(A@B + bias) -------------------
template <int KD, bool RELU, bool OUTH>
__global__ void __launch_bounds__(32)
gemm16(const _Float16* __restrict__ A, const _Float16* __restrict__ Bw,
       const float* __restrict__ bias, void* __restrict__ Out, int ldA, int ldB) {
  const int lane = threadIdx.x;
  const int m0 = blockIdx.y * 16, n0 = blockIdx.x * 16;
  v8f acc = {};
#pragma unroll
  for (int k0 = 0; k0 < KD; k0 += 32) {
    v16h a = frag_a(A, ldA, m0, k0, lane);
    v16h b = frag_b(Bw, ldB, k0, n0, lane);
    acc = __builtin_amdgcn_wmma_f32_16x16x32_f16(false, a, false, b, (short)0, acc,
                                                 false, false);
  }
  const int col = n0 + (lane & 15);
  const int mb  = m0 + ((lane & 16) ? 8 : 0);
  const float bv = bias ? bias[col] : 0.f;
#pragma unroll
  for (int r = 0; r < 8; ++r) {
    float v = acc[r] + bv;
    if (RELU) v = v > 0.f ? v : 0.f;
    if (OUTH) ((_Float16*)Out)[(size_t)(mb + r) * ldB + col] = (_Float16)v;
    else      ((float*)Out)[(size_t)(mb + r) * ldB + col] = v;
  }
}

// ---------------- stage 0: fold W1/mw into 4-wide score weights ----------------
__global__ void __launch_bounds__(32)
prep_cw(const float* __restrict__ W1, const float* __restrict__ b1,
        const float* __restrict__ mw, const float* __restrict__ mb,
        float* __restrict__ cwcb) {
  const int lane = threadIdx.x;
  if (lane < 16) {
    const int n = lane >> 2, d = lane & 3;
    float s = 0.f;
    for (int h = 0; h < DM; ++h) s += W1[d * DM + h] * mw[n * DM + h];
    cwcb[n * 4 + d] = s;
  } else if (lane < 16 + NHD) {
    const int n = lane - 16;
    float s = 0.f;
    for (int h = 0; h < DM; ++h) s += b1[h] * mw[n * DM + h];
    cwcb[16 + n] = s + mb[n];
  }
}

// ---- stage 1: e[n,b,i,j] = exp(leaky(feature·cw + cb)), denom = row sum over j ----
__global__ void __launch_bounds__(32)
edge_scores(const float* __restrict__ feature, const float* __restrict__ cwcb,
            float* __restrict__ e, float* __restrict__ denom) {
  const int row = blockIdx.x;              // (n*B + b)*N + i
  const int nh  = row / (BSZ * NNODE);
  const int bi  = row % (BSZ * NNODE);     // b*N + i
  const int lane = threadIdx.x;
  const float c0 = cwcb[nh * 4 + 0], c1 = cwcb[nh * 4 + 1];
  const float c2 = cwcb[nh * 4 + 2], c3 = cwcb[nh * 4 + 3];
  const float cb = cwcb[16 + nh];
  const float* fb = feature + (size_t)bi * NNODE * DINC;
  float* er = e + (size_t)row * NNODE;
  float s = 0.f;
#pragma unroll
  for (int t = 0; t < NNODE / 32; ++t) {
    const int j = lane + 32 * t;
    const float* f = fb + j * DINC;
    float sc = f[0] * c0 + f[1] * c1 + f[2] * c2 + f[3] * c3 + cb;
    sc = sc >= 0.f ? sc : SLOPE * sc;
    const float ev = __expf(sc);
    er[j] = ev;
    s += ev;
  }
  s = wave_sum(s);
  if (lane == 0) denom[row] = s;
}

// ---- stage 2: g[b,j,d] = sum_i ws*feature[b,i,j,d],  S[b,j] = sum_i ws ----
__global__ void __launch_bounds__(32)
aggregate(const float* __restrict__ feature, const float* __restrict__ e,
          const float* __restrict__ denom, float* __restrict__ g,
          float* __restrict__ Ssum) {
  const int bj = blockIdx.x;               // b*N + j
  const int b = bj >> 8, j = bj & 255;
  const int lane = threadIdx.x;
  float a0 = 0.f, a1 = 0.f, a2 = 0.f, a3 = 0.f, as = 0.f;
  for (int i = lane; i < NNODE; i += 32) {
    float ws = 0.f;
#pragma unroll
    for (int n = 0; n < NHD; ++n) {
      const int r = (n * BSZ + b) * NNODE + i;
      ws += e[(size_t)r * NNODE + j] / denom[r];
    }
    const float* f = feature + ((size_t)(b * NNODE + i) * NNODE + j) * DINC;
    a0 += ws * f[0]; a1 += ws * f[1]; a2 += ws * f[2]; a3 += ws * f[3];
    as += ws;
  }
  a0 = wave_sum(a0); a1 = wave_sum(a1); a2 = wave_sum(a2); a3 = wave_sum(a3);
  as = wave_sum(as);
  if (lane == 0) {
    g[bj * 4 + 0] = a0; g[bj * 4 + 1] = a1; g[bj * 4 + 2] = a2; g[bj * 4 + 3] = a3;
    Ssum[bj] = as;
  }
}

// ---- stage 3: h[b,j,:] = (g@W1 + S*b1) / NH ----
__global__ void __launch_bounds__(64)
compute_h(const float* __restrict__ g, const float* __restrict__ Ssum,
          const float* __restrict__ W1, const float* __restrict__ b1,
          float* __restrict__ h) {
  const int bj = blockIdx.x, c = threadIdx.x;
  const float v = g[bj * 4 + 0] * W1[0 * DM + c] + g[bj * 4 + 1] * W1[1 * DM + c] +
                  g[bj * 4 + 2] * W1[2 * DM + c] + g[bj * 4 + 3] * W1[3 * DM + c] +
                  Ssum[bj] * b1[c];
  h[(size_t)bj * DM + c] = v * (1.0f / NHD);
}

// ---- stage 4: training-mode BatchNorm over (B,N), biased var ----
__global__ void __launch_bounds__(256)
bn_stats(const float* __restrict__ h, float* __restrict__ bnm, float* __restrict__ bnv) {
  const int c = blockIdx.x, t = threadIdx.x;
  float s = 0.f, ss = 0.f;
  for (int r = t; r < ROWS; r += 256) {
    const float v = h[(size_t)r * DM + c];
    s += v; ss += v * v;
  }
  __shared__ float sh[256], sh2[256];
  sh[t] = s; sh2[t] = ss;
  __syncthreads();
  for (int o = 128; o > 0; o >>= 1) {
    if (t < o) { sh[t] += sh[t + o]; sh2[t] += sh2[t + o]; }
    __syncthreads();
  }
  if (t == 0) {
    const float m = sh[0] * (1.0f / ROWS);
    bnm[c] = m;
    bnv[c] = sh2[0] * (1.0f / ROWS) - m * m;
  }
}

__global__ void __launch_bounds__(256)
bn_apply(const float* __restrict__ h, const float* __restrict__ bnm,
         const float* __restrict__ bnv, const float* __restrict__ gam,
         const float* __restrict__ bet, float* __restrict__ x,
         _Float16* __restrict__ xh) {
  const int i = blockIdx.x * 256 + threadIdx.x;
  const int c = i & (DM - 1);
  const float y = gam[c] * (h[i] - bnm[c]) * rsqrtf(bnv[c] + EPSV) + bet[c];
  x[i] = y;
  xh[i] = (_Float16)y;
}

// ---- f32 -> f16 weight conversion ----
__global__ void __launch_bounds__(256)
cvt_f16(const float* __restrict__ src, _Float16* __restrict__ dst, int n) {
  const int i = blockIdx.x * 256 + threadIdx.x;
  if (i < n) dst[i] = (_Float16)src[i];
}

// ---- attention: S = (Q Kᵀ) * scale, per (b,head); K-dim 16 zero-padded to 32 ----
__global__ void __launch_bounds__(32)
attn_qk(const _Float16* __restrict__ Q, const _Float16* __restrict__ Km,
        float* __restrict__ S, float scale) {
  const int lane = threadIdx.x;
  const int bh = blockIdx.z, b = bh / NHD, h = bh % NHD;
  const int q0 = blockIdx.y * 16, n0 = blockIdx.x * 16;
  const _Float16* Qb = Q + (size_t)b * NNODE * DM + h * DHD;
  const _Float16* Kb = Km + (size_t)b * NNODE * DM + h * DHD;
  v16h a = {};
  {
    const int row = q0 + (lane & 15);
    const int kb  = 0;                         // K 0..15 in lanes 0-15; lanes 16-31 stay 0
    const _Float16* p = Qb + (size_t)row * DM + kb;
#pragma unroll
    for (int t = 0; t < 16; ++t) a[t] = (lane & 16) ? (_Float16)0 : p[t];
  }
  v16h bf = {};
  if (!(lane & 16)) {                          // lanes carrying K>=16 stay zero
    const int key = n0 + (lane & 15);
    const _Float16* p = Kb + (size_t)key * DM;
#pragma unroll
    for (int t = 0; t < 16; ++t) bf[t] = p[t]; // Bᵀ column = K row of this key
  }
  v8f acc = {};
  acc = __builtin_amdgcn_wmma_f32_16x16x32_f16(false, a, false, bf, (short)0, acc,
                                               false, false);
  const int col = n0 + (lane & 15);
  const int mb  = q0 + ((lane & 16) ? 8 : 0);
  float* Sp = S + (size_t)bh * NNODE * NNODE;
#pragma unroll
  for (int r = 0; r < 8; ++r) Sp[(size_t)(mb + r) * NNODE + col] = acc[r] * scale;
}

// ---- row softmax (max-subtracted), f16 probs out ----
__global__ void __launch_bounds__(32)
softmax_row(const float* __restrict__ S, _Float16* __restrict__ P) {
  const int row = blockIdx.x, lane = threadIdx.x;
  const float* s = S + (size_t)row * NNODE;
  _Float16* p = P + (size_t)row * NNODE;
  float vals[NNODE / 32];
  float mx = -3.0e38f;
#pragma unroll
  for (int t = 0; t < NNODE / 32; ++t) {
    vals[t] = s[lane + 32 * t];
    mx = fmaxf(mx, vals[t]);
  }
  mx = wave_max(mx);
  float sum = 0.f;
#pragma unroll
  for (int t = 0; t < NNODE / 32; ++t) {
    vals[t] = __expf(vals[t] - mx);
    sum += vals[t];
  }
  sum = wave_sum(sum);
  const float inv = 1.0f / sum;
#pragma unroll
  for (int t = 0; t < NNODE / 32; ++t) p[lane + 32 * t] = (_Float16)(vals[t] * inv);
}

// ---- O_head = probs @ V_head  (M=256, N=16, K=256) ----
__global__ void __launch_bounds__(32)
attn_av(const _Float16* __restrict__ P, const _Float16* __restrict__ V,
        _Float16* __restrict__ O) {
  const int lane = threadIdx.x;
  const int bh = blockIdx.z, b = bh / NHD, h = bh % NHD;
  const int q0 = blockIdx.y * 16;
  const _Float16* Pp = P + (size_t)bh * NNODE * NNODE;
  const _Float16* Vb = V + (size_t)b * NNODE * DM + h * DHD;
  v8f acc = {};
  for (int k0 = 0; k0 < NNODE; k0 += 32) {
    v16h a = frag_a(Pp, NNODE, q0, k0, lane);
    const int col = lane & 15;                 // head dim
    const int kb  = k0 + ((lane & 16) ? 16 : 0);
    v16h bf;
#pragma unroll
    for (int t = 0; t < 16; ++t) bf[t] = Vb[(size_t)(kb + t) * DM + col];
    acc = __builtin_amdgcn_wmma_f32_16x16x32_f16(false, a, false, bf, (short)0, acc,
                                                 false, false);
  }
  const int col = lane & 15;
  const int mb  = q0 + ((lane & 16) ? 8 : 0);
  _Float16* Ob = O + (size_t)b * NNODE * DM + h * DHD;
#pragma unroll
  for (int r = 0; r < 8; ++r) Ob[(size_t)(mb + r) * DM + col] = (_Float16)acc[r];
}

// ---- fused residual add + LayerNorm (one wave per 64-wide token row) ----
__global__ void __launch_bounds__(32)
add_layernorm(const float* __restrict__ xin, const float* __restrict__ proj,
              const float* __restrict__ g, const float* __restrict__ bta,
              float* __restrict__ xout, _Float16* __restrict__ xh) {
  const int row = blockIdx.x, lane = threadIdx.x;
  const float* xr = xin + (size_t)row * DM;
  const float* pr = proj + (size_t)row * DM;
  const float v0 = xr[lane] + pr[lane];
  const float v1 = xr[lane + 32] + pr[lane + 32];
  const float mean = wave_sum(v0 + v1) * (1.0f / DM);
  const float d0 = v0 - mean, d1 = v1 - mean;
  const float var = wave_sum(d0 * d0 + d1 * d1) * (1.0f / DM);
  const float inv = rsqrtf(var + EPSV);
  const float y0 = g[lane] * d0 * inv + bta[lane];
  const float y1 = g[lane + 32] * d1 * inv + bta[lane + 32];
  xout[(size_t)row * DM + lane] = y0;
  xout[(size_t)row * DM + lane + 32] = y1;
  xh[(size_t)row * DM + lane] = (_Float16)y0;
  xh[(size_t)row * DM + lane + 32] = (_Float16)y1;
}

// ------------------------------------------------------------------------------
extern "C" void kernel_launch(void* const* d_in, const int* in_sizes, int n_in,
                              void* d_out, int out_size, void* d_ws, size_t ws_size,
                              hipStream_t stream) {
  (void)in_sizes; (void)n_in; (void)out_size; (void)ws_size;
  const float* feature = (const float*)d_in[0];
  const float* W1   = (const float*)d_in[1];
  const float* b1   = (const float*)d_in[2];
  const float* mw   = (const float*)d_in[3];
  const float* mb   = (const float*)d_in[4];
  const float* bng  = (const float*)d_in[5];
  const float* bnb  = (const float*)d_in[6];
  const float* Wq   = (const float*)d_in[7];
  const float* bq   = (const float*)d_in[8];
  const float* Wk   = (const float*)d_in[9];
  const float* bk   = (const float*)d_in[10];
  const float* Wv   = (const float*)d_in[11];
  const float* bv   = (const float*)d_in[12];
  const float* Wo   = (const float*)d_in[13];
  const float* bo   = (const float*)d_in[14];
  const float* ln1g = (const float*)d_in[15];
  const float* ln1b = (const float*)d_in[16];
  const float* Wff1 = (const float*)d_in[17];
  const float* bff1 = (const float*)d_in[18];
  const float* Wff2 = (const float*)d_in[19];
  const float* bff2 = (const float*)d_in[20];
  const float* ln2g = (const float*)d_in[21];
  const float* ln2b = (const float*)d_in[22];

  char* ws = (char*)d_ws;
  float*     eBuf  = (float*)(ws + OFF_E);      // also attention scores
  _Float16*  probs = (_Float16*)(ws + OFF_PROBS);
  float*     den   = (float*)(ws + OFF_DEN);
  float*     gBuf  = (float*)(ws + OFF_G);
  float*     Ssum  = (float*)(ws + OFF_S);
  float*     hBuf  = (float*)(ws + OFF_H);
  float*     xBuf  = (float*)(ws + OFF_X);
  _Float16*  xh    = (_Float16*)(ws + OFF_XH);
  float*     bnm   = (float*)(ws + OFF_BN);
  float*     bnv   = bnm + DM;
  float*     cwcb  = (float*)(ws + OFF_CW);
  _Float16*  w16   = (_Float16*)(ws + OFF_W16);
  _Float16*  Qh    = (_Float16*)(ws + OFF_Q);
  _Float16*  Kh    = (_Float16*)(ws + OFF_K);
  _Float16*  Vh    = (_Float16*)(ws + OFF_V);
  _Float16*  Oh    = (_Float16*)(ws + OFF_O);
  float*     proj  = (float*)(ws + OFF_PROJ);
  _Float16*  hh    = (_Float16*)(ws + OFF_HH);

  // --- graph attention front-end (DIN=4 folded; no feat materialization) ---
  prep_cw<<<1, 32, 0, stream>>>(W1, b1, mw, mb, cwcb);
  edge_scores<<<NHD * BSZ * NNODE, 32, 0, stream>>>(feature, cwcb, eBuf, den);
  aggregate<<<BSZ * NNODE, 32, 0, stream>>>(feature, eBuf, den, gBuf, Ssum);
  compute_h<<<BSZ * NNODE, 64, 0, stream>>>(gBuf, Ssum, W1, b1, hBuf);
  bn_stats<<<DM, 256, 0, stream>>>(hBuf, bnm, bnv);
  bn_apply<<<ROWS * DM / 256, 256, 0, stream>>>(hBuf, bnm, bnv, bng, bnb, xBuf, xh);

  // --- convert transformer weights to f16 once per launch ---
  cvt_f16<<<(NLAY * DM * DM + 255) / 256, 256, 0, stream>>>(Wq, w16 + WQ16, NLAY * DM * DM);
  cvt_f16<<<(NLAY * DM * DM + 255) / 256, 256, 0, stream>>>(Wk, w16 + WK16, NLAY * DM * DM);
  cvt_f16<<<(NLAY * DM * DM + 255) / 256, 256, 0, stream>>>(Wv, w16 + WV16, NLAY * DM * DM);
  cvt_f16<<<(NLAY * DM * DM + 255) / 256, 256, 0, stream>>>(Wo, w16 + WO16, NLAY * DM * DM);
  cvt_f16<<<(NLAY * DM * 2 * DM + 255) / 256, 256, 0, stream>>>(Wff1, w16 + WF1, NLAY * DM * 2 * DM);
  cvt_f16<<<(NLAY * 2 * DM * DM + 255) / 256, 256, 0, stream>>>(Wff2, w16 + WF2, NLAY * 2 * DM * DM);

  const float scale = 0.25f;  // 1/sqrt(DH=16)
  const dim3 gProj(DM / 16, ROWS / 16);        // (4, 64)
  const dim3 gFF1(2 * DM / 16, ROWS / 16);     // (8, 64)
  const dim3 gQK(NNODE / 16, NNODE / 16, BSZ * NHD);
  const dim3 gAV(1, NNODE / 16, BSZ * NHD);

  for (int l = 0; l < NLAY; ++l) {
    // QKV projections (WMMA, f16 out for attention)
    gemm16<DM, false, true><<<gProj, 32, 0, stream>>>(xh, w16 + WQ16 + l * DM * DM,
                                                      bq + l * DM, Qh, DM, DM);
    gemm16<DM, false, true><<<gProj, 32, 0, stream>>>(xh, w16 + WK16 + l * DM * DM,
                                                      bk + l * DM, Kh, DM, DM);
    gemm16<DM, false, true><<<gProj, 32, 0, stream>>>(xh, w16 + WV16 + l * DM * DM,
                                                      bv + l * DM, Vh, DM, DM);
    // scaled dot-product attention
    attn_qk<<<gQK, 32, 0, stream>>>(Qh, Kh, eBuf, scale);
    softmax_row<<<BSZ * NHD * NNODE, 32, 0, stream>>>(eBuf, probs);
    attn_av<<<gAV, 32, 0, stream>>>(probs, Vh, Oh);
    // output projection + residual + LN1
    gemm16<DM, false, false><<<gProj, 32, 0, stream>>>(Oh, w16 + WO16 + l * DM * DM,
                                                       bo + l * DM, proj, DM, DM);
    add_layernorm<<<ROWS, 32, 0, stream>>>(xBuf, proj, ln1g + l * DM, ln1b + l * DM,
                                           xBuf, xh);
    // feed-forward + residual + LN2
    gemm16<DM, true, true><<<gFF1, 32, 0, stream>>>(xh, w16 + WF1 + l * DM * 2 * DM,
                                                    bff1 + l * 2 * DM, hh, DM, 2 * DM);
    gemm16<2 * DM, false, false><<<gProj, 32, 0, stream>>>(hh, w16 + WF2 + l * 2 * DM * DM,
                                                           bff2 + l * DM, proj, 2 * DM, DM);
    float* xo = (l == NLAY - 1) ? (float*)d_out : xBuf;
    add_layernorm<<<ROWS, 32, 0, stream>>>(xBuf, proj, ln2g + l * DM, ln2b + l * DM,
                                           xo, xh);
  }
}
